// FrequencyTimeAttention_87462714016443
// MI455X (gfx1250) — matrix operand order repaired and checked by reading, not compiled, not run on hardware
//
#include <hip/hip_runtime.h>
#include <hip/hip_bf16.h>

// ---------------------------------------------------------------------------
// FrequencyTimeAttention for MI455X (gfx1250, wave32, WMMA)
//   B=1, DIM=256, F=T=64  -> S=4096 tokens, 8 heads x 32 head_dim
// Pipeline (all bf16 operands, f32 accumulation via v_wmma_f32_16x16x32_bf16):
//   k_convT   : Wqkv/Wout fp32 -> bf16, transposed (row = output column)
//   k_ln<T>   : LayerNorm over C=256 (transposed read for x)
//   k_qkvgemm : (4096x256)@(256x768)+b, q pre-scaled, bf16 out
//   k_flash   : flash attention; online softmax in registers; V chunks
//               double-buffered via GLOBAL_LOAD_ASYNC_TO_LDS + s_wait_asynccnt
//   k_ln<F>   : LayerNorm of attention output
//   k_outgemm : (4096x256)@(256x256)+b + residual, stores transposed (c,s)
// ---------------------------------------------------------------------------

typedef __bf16 bf16;
typedef __attribute__((ext_vector_type(16))) __bf16 v16bf;
typedef __attribute__((ext_vector_type(8)))  __bf16 v8bf;
typedef __attribute__((ext_vector_type(8)))  float  v8f;
typedef int v4i __attribute__((vector_size(16)));   // matches async-LDS builtin param

#define S_TOT 4096
#define CDIM  256
#define HEADS 8
#define HDIM  32
#define QKVW  768
#define ATT_SCALE 0.17677669529663687f  // 32^-0.5
#define LN_EPS 1e-5f

#define AS1 __attribute__((address_space(1)))
#define AS3 __attribute__((address_space(3)))

#if defined(__has_builtin)
#if __has_builtin(__builtin_amdgcn_global_load_async_to_lds_b128)
#define HAVE_ASYNC_LDS 1
#endif
#endif
#ifndef HAVE_ASYNC_LDS
#define HAVE_ASYNC_LDS 0
#endif

#if HAVE_ASYNC_LDS
#if __has_builtin(__builtin_amdgcn_s_wait_asynccnt)
#define WAIT_ASYNC(n) __builtin_amdgcn_s_wait_asynccnt(n)
#else
#define WAIT_ASYNC(n) asm volatile("s_wait_asynccnt " #n ::: "memory")
#endif
// global (addrspace 1) int4* from a const bf16*, LDS (addrspace 3) int4*
#define GPTR(p) ((AS1 v4i*)(v4i*)(bf16*)(p))
#define LPTR(p) ((AS3 v4i*)(v4i*)(p))
#endif

union V16U { v16bf v; v8bf h[2]; };

// A operand (16x32 bf16): lane = row M=lane&15; lo lanes K 0..15, hi 16..31
// (sequential). p must already include the (hi?16:0) K offset.
__device__ __forceinline__ v16bf loadA16(const bf16* p) {
    V16U u;
    u.h[0] = *(const v8bf*)(p);
    u.h[1] = *(const v8bf*)(p + 8);
    return u.v;
}
// B operand (32x16 bf16): lane = column N=lane&15; lo lanes K 0..15, hi 16..31
// (sequential). p must already include the (hi?16:0) K offset.
__device__ __forceinline__ v16bf loadB16(const bf16* p) {
    V16U u;
    u.h[0] = *(const v8bf*)(p);
    u.h[1] = *(const v8bf*)(p + 8);
    return u.v;
}

__device__ __forceinline__ v8f wmma_bf16(v16bf a, v16bf b, v8f c) {
    return __builtin_amdgcn_wmma_f32_16x16x32_bf16(false, a, false, b,
                                                   (short)0, c, false, false);
}

// ---------------------------------------------------------------------------
// Transposed weight conversion: dst[j*K + k] = (bf16) src[k*J + j]
// ---------------------------------------------------------------------------
__global__ void k_convT(const float* __restrict__ src, bf16* __restrict__ dst,
                        int K, int J) {
    int i = blockIdx.x * blockDim.x + threadIdx.x;
    if (i >= J * K) return;
    int j = i / K, k = i - j * K;
    dst[j * K + k] = (bf16)src[k * J + j];
}

// ---------------------------------------------------------------------------
// LayerNorm over C=256. TRANS=true: in is (C, S) (the raw x); else (S, C).
// ---------------------------------------------------------------------------
template <bool TRANS>
__global__ __launch_bounds__(CDIM)
void k_ln(const float* __restrict__ in, const float* __restrict__ g,
          const float* __restrict__ b, bf16* __restrict__ out) {
    const int s = blockIdx.x;
    const int c = threadIdx.x;
    float v = TRANS ? in[c * S_TOT + s] : in[s * CDIM + c];
    __shared__ float r1[CDIM];
    __shared__ float r2[CDIM];
    r1[c] = v;
    r2[c] = v * v;
    __syncthreads();
    for (int off = CDIM / 2; off > 0; off >>= 1) {
        if (c < off) { r1[c] += r1[c + off]; r2[c] += r2[c + off]; }
        __syncthreads();
    }
    float mean = r1[0] * (1.0f / CDIM);
    float var  = r2[0] * (1.0f / CDIM) - mean * mean;
    float inv  = rsqrtf(var + LN_EPS);
    out[s * CDIM + c] = (bf16)((v - mean) * inv * g[c] + b[c]);
}

// ---------------------------------------------------------------------------
// QKV GEMM: qkv(S x 768) = xn(S x 256) @ Wqkv + bqkv.  Wt is (768 x 256)
// (transposed, bf16).  q columns (j < 256) pre-multiplied by ATT_SCALE.
// Wave = 16(M) x 32(N) tile; 8 K-steps of 32.
// ---------------------------------------------------------------------------
__global__ __launch_bounds__(128)
void k_qkvgemm(const bf16* __restrict__ xn, const bf16* __restrict__ Wt,
               const float* __restrict__ bias, bf16* __restrict__ qkv) {
    const int lane = threadIdx.x & 31;
    const int wave = threadIdx.x >> 5;
    const int widx = blockIdx.x * 4 + wave;      // 256 * 24 waves
    const int m0 = (widx / 24) * 16;
    const int n0 = (widx % 24) * 32;
    const int rowA  = m0 + (lane & 15);
    const int kA    = (lane & 16) ? 16 : 0;
    const int col0  = n0 + (lane & 15);
    const int kB    = (lane & 16) ? 16 : 0;

    v8f c0 = {}, c1 = {};
    for (int k0 = 0; k0 < CDIM; k0 += 32) {
        v16bf a  = loadA16(xn + rowA * CDIM + k0 + kA);
        v16bf b0 = loadB16(Wt + (col0)*CDIM + k0 + kB);
        v16bf b1 = loadB16(Wt + (col0 + 16) * CDIM + k0 + kB);
        c0 = wmma_bf16(a, b0, c0);
        c1 = wmma_bf16(a, b1, c1);
    }
    const int roff = (lane & 16) ? 8 : 0;
    const float bias0 = bias[col0];
    const float bias1 = bias[col0 + 16];
    const float sc = (n0 < CDIM) ? ATT_SCALE : 1.0f;  // whole 32-tile is q/k/v
#pragma unroll
    for (int r = 0; r < 8; ++r) {
        const int row = m0 + roff + r;
        qkv[row * QKVW + col0]      = (bf16)((c0[r] + bias0) * sc);
        qkv[row * QKVW + col0 + 16] = (bf16)((c1[r] + bias1) * sc);
    }
}

// ---------------------------------------------------------------------------
// Flash attention. Block = 4 waves = 64 queries of one head; each wave owns a
// 16-query tile. Iterates over keys in chunks of 32:
//   S(16x32) = Q(16x32) Kᵀ  (2 wmma; K's B operand is contiguous in memory)
//   online softmax (row stats across lanes via shfl_xor within half-wave)
//   P bf16 via per-wave LDS (C-layout -> A-layout)
//   O(16x32) += P(16x32) V(32x32) (2 wmma)
// V chunks are copied global->LDS with async-to-LDS, double buffered so chunk
// i+1 is in flight while chunk i is transposed+consumed; each thread reads
// back only its own 16B, so per-wave s_wait_asynccnt is the only sync needed
// for Vraw.
// ---------------------------------------------------------------------------
__global__ __launch_bounds__(128)
void k_flash(const bf16* __restrict__ qkv, float* __restrict__ Obuf) {
    const int lane  = threadIdx.x & 31;
    const int wave  = threadIdx.x >> 5;
    const int h     = blockIdx.x >> 6;           // 8 heads
    const int qb    = blockIdx.x & 63;           // 64 query blocks
    const int m0    = qb * 64 + wave * 16;
    const int row16 = lane & 15;
    const int hi    = (lane & 16) ? 1 : 0;

    __shared__ __align__(16) bf16 Vt[32 * 40];       // V chunk, transposed [d][k], padded
    __shared__ __align__(16) bf16 Plds[4][16 * 32];  // per-wave P tile
#if HAVE_ASYNC_LDS
    __shared__ __align__(16) bf16 Vraw[2][32 * 32];  // async landing pads
#endif

    // Q tile (A layout), scale already folded in at QKV store time
    const v16bf A = loadA16(qkv + (m0 + row16) * QKVW + h * HDIM + (hi ? 16 : 0));

    v8f O0 = {}, O1 = {};
    float mrow[8], lrow[8];
#pragma unroll
    for (int r = 0; r < 8; ++r) { mrow[r] = -__builtin_inff(); lrow[r] = 0.0f; }

    const int tid  = threadIdx.x;
    const int vk   = tid >> 2;        // key row in chunk: 0..31
    const int vseg = (tid & 3) * 8;   // head-dim segment
    const bf16* vsrc = qkv + 2 * CDIM + h * HDIM + vseg;  // + (kc+vk)*QKVW

#if HAVE_ASYNC_LDS
    __builtin_amdgcn_global_load_async_to_lds_b128(
        GPTR(vsrc + vk * QKVW), LPTR(&Vraw[0][vk * 32 + vseg]), 0, 0);
#endif

    for (int kc = 0; kc < S_TOT; kc += 32) {
#if HAVE_ASYNC_LDS
        const int buf = (kc >> 5) & 1;
        if (kc + 32 < S_TOT) {
            __builtin_amdgcn_global_load_async_to_lds_b128(
                GPTR(vsrc + (kc + 32 + vk) * QKVW),
                LPTR(&Vraw[buf ^ 1][vk * 32 + vseg]), 0, 0);
            WAIT_ASYNC(1);   // chunk kc landed, chunk kc+32 still in flight
        } else {
            WAIT_ASYNC(0);
        }
#endif
        // prefetch next chunk's K rows (speculative; stays inside workspace)
        __builtin_prefetch(qkv + (kc + 32 + row16) * QKVW + CDIM + h * HDIM, 0, 1);

        __syncthreads();  // previous chunk's Vt readers done
        {   // staging: Vt[d][k] = V[kc+k][d]
#if HAVE_ASYNC_LDS
            v8bf vv = *(const v8bf*)&Vraw[buf][vk * 32 + vseg];
#else
            v8bf vv = *(const v8bf*)(vsrc + (kc + vk) * QKVW);
#endif
#pragma unroll
            for (int i = 0; i < 8; ++i) Vt[(vseg + i) * 40 + vk] = vv[i];
        }
        __syncthreads();

        // S = Q Kᵀ : B column n = key row -> contiguous 16 bf16 per lane
        v16bf B0 = loadB16(qkv + (kc + row16) * QKVW + CDIM + h * HDIM + (hi ? 16 : 0));
        v16bf B1 = loadB16(qkv + (kc + 16 + row16) * QKVW + CDIM + h * HDIM + (hi ? 16 : 0));
        v8f S0 = {}, S1 = {};
        S0 = wmma_bf16(A, B0, S0);
        S1 = wmma_bf16(A, B1, S1);

        // online softmax; each half-wave's 16 lanes hold one row set (r / r+8)
#pragma unroll
        for (int r = 0; r < 8; ++r) {
            float s0 = S0[r], s1 = S1[r];
            float cm = fmaxf(s0, s1);
            cm = fmaxf(cm, __shfl_xor(cm, 1));
            cm = fmaxf(cm, __shfl_xor(cm, 2));
            cm = fmaxf(cm, __shfl_xor(cm, 4));
            cm = fmaxf(cm, __shfl_xor(cm, 8));
            float mn    = fmaxf(mrow[r], cm);
            float alpha = __expf(mrow[r] - mn);
            float p0    = __expf(s0 - mn);
            float p1    = __expf(s1 - mn);
            float ps    = p0 + p1;
            ps += __shfl_xor(ps, 1);
            ps += __shfl_xor(ps, 2);
            ps += __shfl_xor(ps, 4);
            ps += __shfl_xor(ps, 8);
            lrow[r] = lrow[r] * alpha + ps;
            mrow[r] = mn;
            O0[r] *= alpha;
            O1[r] *= alpha;
            const int prow = r + 8 * hi;
            Plds[wave][prow * 32 + row16]      = (bf16)p0;
            Plds[wave][prow * 32 + row16 + 16] = (bf16)p1;
        }
        // same-wave DS ops are in order: reload P in A layout
        v16bf Pa  = loadA16(&Plds[wave][row16 * 32 + (hi ? 16 : 0)]);
        v16bf Vb0 = loadB16(&Vt[(row16) * 40 + (hi ? 16 : 0)]);
        v16bf Vb1 = loadB16(&Vt[(row16 + 16) * 40 + (hi ? 16 : 0)]);
        O0 = wmma_bf16(Pa, Vb0, O0);
        O1 = wmma_bf16(Pa, Vb1, O1);
    }

#pragma unroll
    for (int r = 0; r < 8; ++r) {
        const float inv = 1.0f / lrow[r];
        const int srow = m0 + 8 * hi + r;
        Obuf[srow * CDIM + h * HDIM + row16]      = O0[r] * inv;
        Obuf[srow * CDIM + h * HDIM + row16 + 16] = O1[r] * inv;
    }
}

// ---------------------------------------------------------------------------
// Output GEMM + bias + residual, stored transposed: out[c][s].
// C layout (lane = column c, VGPR r = consecutive s rows) -> float4 stores.
// ---------------------------------------------------------------------------
__global__ __launch_bounds__(128)
void k_outgemm(const bf16* __restrict__ on, const bf16* __restrict__ Wt,
               const float* __restrict__ bout, const float* __restrict__ x,
               float* __restrict__ out) {
    const int lane = threadIdx.x & 31;
    const int wave = threadIdx.x >> 5;
    const int widx = blockIdx.x * 4 + wave;      // 256 * 8 waves
    const int m0 = (widx >> 3) * 16;
    const int n0 = (widx & 7) * 32;
    const int rowA = m0 + (lane & 15);
    const int kA   = (lane & 16) ? 16 : 0;
    const int col0 = n0 + (lane & 15);
    const int kB   = (lane & 16) ? 16 : 0;

    v8f c0 = {}, c1 = {};
    for (int k0 = 0; k0 < CDIM; k0 += 32) {
        v16bf a  = loadA16(on + rowA * CDIM + k0 + kA);
        v16bf b0 = loadB16(Wt + (col0)*CDIM + k0 + kB);
        v16bf b1 = loadB16(Wt + (col0 + 16) * CDIM + k0 + kB);
        c0 = wmma_bf16(a, b0, c0);
        c1 = wmma_bf16(a, b1, c1);
    }
    const int roff = (lane & 16) ? 8 : 0;
    const float bb0 = bout[col0];
    const float bb1 = bout[col0 + 16];
    const long base0 = (long)(col0)*S_TOT + m0 + roff;
    const long base1 = (long)(col0 + 16) * S_TOT + m0 + roff;

    float y0[8], y1[8];
#pragma unroll
    for (int r = 0; r < 8; ++r) {
        y0[r] = c0[r] + bb0 + x[base0 + r];  // residual = permuted input
        y1[r] = c1[r] + bb1 + x[base1 + r];
    }
    float4* p0 = (float4*)(out + base0);
    float4* p1 = (float4*)(out + base1);
    p0[0] = make_float4(y0[0], y0[1], y0[2], y0[3]);
    p0[1] = make_float4(y0[4], y0[5], y0[6], y0[7]);
    p1[0] = make_float4(y1[0], y1[1], y1[2], y1[3]);
    p1[1] = make_float4(y1[4], y1[5], y1[6], y1[7]);
}

// ---------------------------------------------------------------------------
extern "C" void kernel_launch(void* const* d_in, const int* in_sizes, int n_in,
                              void* d_out, int out_size, void* d_ws, size_t ws_size,
                              hipStream_t stream) {
    (void)in_sizes; (void)n_in; (void)out_size; (void)ws_size;
    const float* x    = (const float*)d_in[0];
    const float* Wqkv = (const float*)d_in[1];
    const float* bqkv = (const float*)d_in[2];
    const float* Wout = (const float*)d_in[3];
    const float* bout = (const float*)d_in[4];
    const float* g1   = (const float*)d_in[5];
    const float* b1   = (const float*)d_in[6];
    const float* g2   = (const float*)d_in[7];
    const float* b2   = (const float*)d_in[8];
    float* out = (float*)d_out;

    char* ws = (char*)d_ws;
    bf16*  qkv    = (bf16*)(ws);                       // 4096*768*2  = 6291456
    float* Obuf   = (float*)(ws + 6291456);            // 4096*256*4  = 4194304
    bf16*  Wt_qkv = (bf16*)(ws + 10485760);            // 768*256*2   = 393216
    bf16*  Wt_out = (bf16*)(ws + 10878976);            // 256*256*2   = 131072
    bf16*  xn     = (bf16*)(ws + 11010048);            // 4096*256*2  = 2097152
    bf16*  onrm   = (bf16*)(ws + 13107200);            // 4096*256*2  = 2097152

    k_convT<<<(768 * 256 + 255) / 256, 256, 0, stream>>>(Wqkv, Wt_qkv, 256, 768);
    k_convT<<<(256 * 256 + 255) / 256, 256, 0, stream>>>(Wout, Wt_out, 256, 256);
    k_ln<true><<<S_TOT, CDIM, 0, stream>>>(x, g1, b1, xn);
    k_qkvgemm<<<1536, 128, 0, stream>>>(xn, Wt_qkv, bqkv, qkv);
    k_flash<<<512, 128, 0, stream>>>(qkv, Obuf);
    k_ln<false><<<S_TOT, CDIM, 0, stream>>>(Obuf, g2, b2, onrm);
    k_outgemm<<<512, 128, 0, stream>>>(onrm, Wt_out, bout, x, out);
}
